// EdgeModel_49830210568747
// MI455X (gfx1250) — matrix operand order repaired and checked by reading, not compiled, not run on hardware
//
#include <hip/hip_runtime.h>

// ---------------------------------------------------------------------------
// Fused EdgeModel MLP for MI455X (gfx1250, wave32, WMMA).
// out[E,64] = relu(concat(src,dest,edge,u[batch]) @ W1 + b1) @ W2 + b2
// f16 WMMA (v_wmma_f32_16x16x32_f16) with f32 accumulation; weights resident
// in LDS (transposed, padded); per-wave 16-edge tiles.
// kb-outer / nb-inner loop order keeps live VGPRs < 256 (no s_set_vgpr_msb).
// ---------------------------------------------------------------------------

typedef _Float16 h4   __attribute__((ext_vector_type(4)));
typedef _Float16 v8h  __attribute__((ext_vector_type(8)));
typedef _Float16 v16h __attribute__((ext_vector_type(16)));
typedef float    v8f  __attribute__((ext_vector_type(8)));

#define D_NODE   64
#define D_IN     256   // 2*64 + 64 + 64
#define HIDDEN   128
#define OUT_DIM  64
#define WAVES_PER_BLOCK 4

// padded LDS row strides (in f16 elements) -> 528B / 272B byte strides,
// both 16B-aligned and == 4 banks (mod 64) so b128 reads are conflict-light.
#define A_STRIDE  264   // 256 + 8
#define H_STRIDE  136   // 128 + 8
#define W1_STRIDE 264
#define W2_STRIDE 136

__device__ __forceinline__ int imin(int a, int b) { return a < b ? a : b; }

__global__ __launch_bounds__(128)
void edge_mlp_kernel(const float* __restrict__ src_x,
                     const float* __restrict__ dest_x,
                     const float* __restrict__ edge_attr,
                     const float* __restrict__ u,
                     const int*   __restrict__ batch,
                     const float* __restrict__ W1,
                     const float* __restrict__ b1,
                     const float* __restrict__ W2,
                     const float* __restrict__ b2,
                     float*       __restrict__ out,
                     int E)
{
    // LDS: f16 transposed weights + per-wave staging tiles. ~133 KB total
    // (2 workgroups co-resident per 320 KB WGP).
    __shared__ __align__(16) _Float16 sW1T[HIDDEN  * W1_STRIDE];           // [n][k] 67584 B
    __shared__ __align__(16) _Float16 sW2T[OUT_DIM * W2_STRIDE];           // [n][k] 17408 B
    __shared__ __align__(16) _Float16 sA[WAVES_PER_BLOCK * 16 * A_STRIDE]; // 33792 B
    __shared__ __align__(16) _Float16 sH[WAVES_PER_BLOCK * 16 * H_STRIDE]; // 17408 B

    const int tid  = threadIdx.x;
    const int lane = tid & 31;
    const int wave = tid >> 5;
    const int m    = lane & 15;   // row within 16-tile (A) / col N (B,C)
    const int hi   = lane >> 4;   // half-wave select

    // ---- Stage transposed f16 weights into LDS (coalesced global reads) ----
    for (int i = tid; i < D_IN * HIDDEN; i += 128) {
        int k = i >> 7;         // 0..255
        int n = i & 127;        // 0..127
        sW1T[n * W1_STRIDE + k] = (_Float16)W1[i];
    }
    for (int i = tid; i < HIDDEN * OUT_DIM; i += 128) {
        int k = i >> 6;         // 0..127
        int n = i & 63;         // 0..63
        sW2T[n * W2_STRIDE + k] = (_Float16)W2[i];
    }

    // Per-lane bias registers: lane holds column n = nb*16 + m of C tile.
    float rb1[8], rb2[4];
#pragma unroll
    for (int nb = 0; nb < 8; ++nb) rb1[nb] = b1[nb * 16 + m];
#pragma unroll
    for (int nb = 0; nb < 4; ++nb) rb2[nb] = b2[nb * 16 + m];

    __syncthreads();

    _Float16* Aw = sA + wave * 16 * A_STRIDE;
    _Float16* Hw = sH + wave * 16 * H_STRIDE;

    const int tiles    = (E + 15) >> 4;
    const int gwave    = blockIdx.x * WAVES_PER_BLOCK + wave;
    const int numWaves = gridDim.x * WAVES_PER_BLOCK;

    for (int t = gwave; t < tiles; t += numWaves) {
        const int base = t << 4;

        // ---- Stage combined [16 x 256] tile into LDS as f16 ----
        // lanes 0-15 handle row 2r (cols 4m..4m+3), lanes 16-31 row 2r+1.
        int g0 = 0;
        if (lane < 16) g0 = batch[imin(base + lane, E - 1)];

        for (int r = 0; r < 8; ++r) {
            const int row = 2 * r + hi;
            const int e   = imin(base + row, E - 1);
            const int g   = __shfl(g0, row, 32);
            const float4 s  = *(const float4*)(src_x     + (size_t)e * D_NODE + 4 * m);
            const float4 d  = *(const float4*)(dest_x    + (size_t)e * D_NODE + 4 * m);
            const float4 ea = *(const float4*)(edge_attr + (size_t)e * D_NODE + 4 * m);
            const float4 uu = *(const float4*)(u         + (size_t)g * D_NODE + 4 * m);
            _Float16* rp = Aw + row * A_STRIDE + 4 * m;
            *(h4*)(rp +   0) = h4{(_Float16)s.x,  (_Float16)s.y,  (_Float16)s.z,  (_Float16)s.w};
            *(h4*)(rp +  64) = h4{(_Float16)d.x,  (_Float16)d.y,  (_Float16)d.z,  (_Float16)d.w};
            *(h4*)(rp + 128) = h4{(_Float16)ea.x, (_Float16)ea.y, (_Float16)ea.z, (_Float16)ea.w};
            *(h4*)(rp + 192) = h4{(_Float16)uu.x, (_Float16)uu.y, (_Float16)uu.z, (_Float16)uu.w};
        }

        // ---- Layer 1: [16x256] @ [256x128] -> relu -> h in LDS (f16) ----
        const _Float16* Arow = Aw + m * A_STRIDE;
        const _Float16* B1row = sW1T + m * W1_STRIDE;   // + nb*16*W1_STRIDE via imm
        v8f acc1[8];
#pragma unroll
        for (int nb = 0; nb < 8; ++nb) acc1[nb] = (v8f){};
#pragma unroll
        for (int kb = 0; kb < 8; ++kb) {
            v16h a;
            // A 16x32 f16 layout: lane=m, K runs [hi*8, +8) and [16+hi*8, +8)
            *((v8h*)&a)     = *(const v8h*)(Arow + kb * 32 + hi * 8);
            *((v8h*)&a + 1) = *(const v8h*)(Arow + kb * 32 + 16 + hi * 8);
#pragma unroll
            for (int nb = 0; nb < 8; ++nb) {
                v16h b;
                // B 32x16 f16 layout: lane=col n, K runs [hi*8, +8) and [16+hi*8, +8)
                const _Float16* bp = B1row + nb * 16 * W1_STRIDE + kb * 32 + hi * 8;
                *((v8h*)&b)     = *(const v8h*)(bp);
                *((v8h*)&b + 1) = *(const v8h*)(bp + 16);
                acc1[nb] = __builtin_amdgcn_wmma_f32_16x16x32_f16(
                               false, a, false, b, (short)0, acc1[nb], false, false);
            }
        }
#pragma unroll
        for (int nb = 0; nb < 8; ++nb) {
            const float bias = rb1[nb];
#pragma unroll
            for (int v = 0; v < 8; ++v) {
                float x = acc1[nb][v] + bias;
                x = x > 0.0f ? x : 0.0f;   // ReLU
                // C layout: row M = v + 8*hi, col N = nb*16 + m
                Hw[(v + 8 * hi) * H_STRIDE + nb * 16 + m] = (_Float16)x;
            }
        }

        // ---- Layer 2: [16x128] @ [128x64] + b2 -> global out ----
        const _Float16* Hrow  = Hw + m * H_STRIDE;
        const _Float16* B2row = sW2T + m * W2_STRIDE;
        v8f acc2[4];
#pragma unroll
        for (int nb = 0; nb < 4; ++nb) acc2[nb] = (v8f){};
#pragma unroll
        for (int kb = 0; kb < 4; ++kb) {
            v16h a;
            *((v8h*)&a)     = *(const v8h*)(Hrow + kb * 32 + hi * 8);
            *((v8h*)&a + 1) = *(const v8h*)(Hrow + kb * 32 + 16 + hi * 8);
#pragma unroll
            for (int nb = 0; nb < 4; ++nb) {
                v16h b;
                const _Float16* bp = B2row + nb * 16 * W2_STRIDE + kb * 32 + hi * 8;
                *((v8h*)&b)     = *(const v8h*)(bp);
                *((v8h*)&b + 1) = *(const v8h*)(bp + 16);
                acc2[nb] = __builtin_amdgcn_wmma_f32_16x16x32_f16(
                               false, a, false, b, (short)0, acc2[nb], false, false);
            }
        }
#pragma unroll
        for (int nb = 0; nb < 4; ++nb) {
            const float bias = rb2[nb];
#pragma unroll
            for (int v = 0; v < 8; ++v) {
                const int e = base + v + 8 * hi;       // C layout row
                if (e < E)
                    out[(size_t)e * OUT_DIM + nb * 16 + m] = acc2[nb][v] + bias;
            }
        }
    }
}

extern "C" void kernel_launch(void* const* d_in, const int* in_sizes, int n_in,
                              void* d_out, int out_size, void* d_ws, size_t ws_size,
                              hipStream_t stream) {
    const float* src_x     = (const float*)d_in[0];
    const float* dest_x    = (const float*)d_in[1];
    const float* edge_attr = (const float*)d_in[2];
    const float* u         = (const float*)d_in[3];
    const int*   batch     = (const int*)  d_in[4];
    const float* W1        = (const float*)d_in[5];
    const float* b1        = (const float*)d_in[6];
    const float* W2        = (const float*)d_in[7];
    const float* b2        = (const float*)d_in[8];
    float*       out       = (float*)d_out;

    const int E     = in_sizes[0] / D_NODE;   // 1,000,000
    const int tiles = (E + 15) / 16;          // 62,500

    int blocks = 1024;                        // persistent grid-stride over tiles
    int maxBlocks = (tiles + WAVES_PER_BLOCK - 1) / WAVES_PER_BLOCK;
    if (blocks > maxBlocks) blocks = maxBlocks;

    edge_mlp_kernel<<<blocks, 128, 0, stream>>>(
        src_x, dest_x, edge_attr, u, batch, W1, b1, W2, b2, out, E);
}
